// my_GCN_61718680043850
// MI455X (gfx1250) — compile-verified, matrix operand order checked by reading, not run on hardware
//
#include <hip/hip_runtime.h>
#include <cmath>

// ---------------- problem constants ----------------
#define NUTT   3000      // utterances per modality
#define N3     9000      // 3 * NUTT (graph nodes)
#define MP     9008      // N3 padded to 16 rows (563 tiles of 16)
#define CDIM   256       // NDIM == NHID
#define NDIAL  100
#define MAXL   30
#define NSPK   9
#define NLAYER 8

typedef _Float16 v16h  __attribute__((ext_vector_type(16)));
typedef _Float16 half8 __attribute__((ext_vector_type(8)));
typedef float    v8f   __attribute__((ext_vector_type(8)));

__device__ __forceinline__ float angf(float c) {
  // 1 - arccos(c * SCALE) / pi
  return 1.0f - acosf(c * 0.99999f) * 0.3183098861837907f;
}

__device__ __forceinline__ float block_sum_256(float v, float* red, int c) {
  red[c] = v; __syncthreads();
  for (int off = 128; off > 0; off >>= 1) {
    if (c < off) red[c] += red[c + off];
    __syncthreads();
  }
  float r = red[0]; __syncthreads();
  return r;
}

// ---------------- 1) weight conversion: f32 -> f16 column-major ----------------
__global__ __launch_bounds__(256) void k_wconv(const float* __restrict__ fc_w,
                                               const float* __restrict__ conv_w,
                                               _Float16* __restrict__ fcWc,
                                               _Float16* __restrict__ convWc) {
  int idx = blockIdx.x * 256 + threadIdx.x;
  if (idx < 256 * 256) {               // fc_w[k][n] -> fcWc[n*256 + k]
    int k = idx >> 8, n = idx & 255;
    fcWc[n * 256 + k] = (_Float16)fc_w[idx];
  }
  const int tot = NLAYER * 512 * 256;  // conv_w[l][k][n] -> convWc[(l*256+n)*512 + k]
  for (int e = idx; e < tot; e += gridDim.x * 256) {
    int l = e / (512 * 256);
    int rr = e % (512 * 256);
    int k = rr >> 8, n = rr & 255;
    convWc[((size_t)l * 256 + n) * 512 + k] = (_Float16)conv_w[e];
  }
}

// ---------------- 2) preprocess: speaker add, x, hats, x-f16 ----------------
__global__ __launch_bounds__(256) void k_pre(const float* __restrict__ a,
                                             const float* __restrict__ v,
                                             const float* __restrict__ t,
                                             const float* __restrict__ speaker,
                                             const int* __restrict__ utt,
                                             const float* __restrict__ spk_table,
                                             float* __restrict__ xf,
                                             float* __restrict__ hats,
                                             _Float16* __restrict__ Xh) {
  __shared__ float red[256];
  __shared__ int s_spk;
  int i = blockIdx.x, c = threadIdx.x;
  if (c == 0) {
    int cum = 0, seg = 0, pos = 0;
    for (int d = 0; d < NDIAL; ++d) { int nx = cum + utt[d]; if (i < nx) { seg = d; pos = i - cum; break; } cum = nx; }
    const float* sp = speaker + ((size_t)pos * NDIAL + seg) * NSPK;
    int best = 0; float bv = sp[0];
    for (int s = 1; s < NSPK; ++s) if (sp[s] > bv) { bv = sp[s]; best = s; }
    s_spk = best;
  }
  __syncthreads();
  float av = a[(size_t)i * CDIM + c];
  float vv = v[(size_t)i * CDIM + c];
  float tv = t[(size_t)i * CDIM + c] + spk_table[(size_t)s_spk * CDIM + c];
  float sa = block_sum_256(av * av, red, c);
  float sv = block_sum_256(vv * vv, red, c);
  float st = block_sum_256(tv * tv, red, c);
  float ra = rsqrtf(sa), rv = rsqrtf(sv), rt = rsqrtf(st);
  size_t ia = (size_t)i * CDIM + c;
  size_t iv = ((size_t)NUTT + i) * CDIM + c;
  size_t it = ((size_t)2 * NUTT + i) * CDIM + c;
  xf[ia] = av; xf[iv] = vv; xf[it] = tv;
  Xh[ia] = (_Float16)av; Xh[iv] = (_Float16)vv; Xh[it] = (_Float16)tv;
  hats[ia] = av * ra; hats[iv] = vv * rv; hats[it] = tv * rt;
}

// ---------------- 3) cross-modal angular similarities (diag links) ----------------
__global__ __launch_bounds__(256) void k_cross(const float* __restrict__ hats,
                                               float* __restrict__ crossv) {
  __shared__ float red[256];
  int i = blockIdx.x, c = threadIdx.x;
  float ha = hats[(size_t)i * CDIM + c];
  float hv = hats[((size_t)NUTT + i) * CDIM + c];
  float ht = hats[((size_t)2 * NUTT + i) * CDIM + c];
  float dav = block_sum_256(ha * hv, red, c);
  float dat = block_sum_256(ha * ht, red, c);
  float dvt = block_sum_256(hv * ht, red, c);
  if (c == 0) {
    crossv[0 * NUTT + i] = angf(dav);
    crossv[1 * NUTT + i] = angf(dat);
    crossv[2 * NUTT + i] = angf(dvt);
  }
}

// ---------------- 4) per-dialogue 30x30 angular blocks ----------------
__global__ __launch_bounds__(256) void k_blocks(const float* __restrict__ hats,
                                                const int* __restrict__ utt,
                                                float* __restrict__ Ablk) {
  __shared__ float sh[MAXL * CDIM];   // 30KB
  __shared__ int s_start, s_L;
  int b = blockIdx.x, m = b / NDIAL, d = b % NDIAL, c = threadIdx.x;
  if (c == 0) { int st = 0; for (int q = 0; q < d; ++q) st += utt[q]; s_start = st; s_L = utt[d]; }
  __syncthreads();
  int start = s_start, L = s_L;
  for (int q = 0; q < L; ++q)
    sh[q * CDIM + c] = hats[((size_t)(m * NUTT + start + q)) * CDIM + c];
  __syncthreads();
  for (int e = c; e < MAXL * MAXL; e += 256) {
    int p = e / MAXL, q = e % MAXL;
    float val = 0.0f;
    if (p < L && q < L) {
      float dt = 0.0f;
      for (int k = 0; k < CDIM; ++k) dt += sh[p * CDIM + k] * sh[q * CDIM + k];
      val = angf(dt);
    }
    Ablk[(size_t)b * (MAXL * MAXL) + e] = val;
  }
}

// ---------------- 5) degree -> D^-1/2 ----------------
__global__ __launch_bounds__(256) void k_dinv(const float* __restrict__ Ablk,
                                              const float* __restrict__ crossv,
                                              const int* __restrict__ utt,
                                              float* __restrict__ dinv) {
  int r = blockIdx.x * 256 + threadIdx.x;
  if (r >= N3) return;
  int m = r / NUTT, i = r % NUTT;
  int cum = 0, d = 0, p = 0;
  for (int q = 0; q < NDIAL; ++q) { int nx = cum + utt[q]; if (i < nx) { d = q; p = i - cum; break; } cum = nx; }
  int L = utt[d];
  const float* row = Ablk + ((size_t)(m * NDIAL + d)) * (MAXL * MAXL) + p * MAXL;
  float s = 0.0f;
  for (int q = 0; q < L; ++q) s += row[q];
  if (m == 0)      s += crossv[0 * NUTT + i] + crossv[1 * NUTT + i];
  else if (m == 1) s += crossv[0 * NUTT + i] + crossv[2 * NUTT + i];
  else             s += crossv[1 * NUTT + i] + crossv[2 * NUTT + i];
  dinv[r] = rsqrtf(s);
}

// ---------------- 6) normalize blocks ----------------
__global__ __launch_bounds__(256) void k_normA(float* __restrict__ Ablk,
                                               const float* __restrict__ dinv,
                                               const int* __restrict__ utt) {
  __shared__ int s_start, s_L;
  int b = blockIdx.x, m = b / NDIAL, d = b % NDIAL, c = threadIdx.x;
  if (c == 0) { int st = 0; for (int q = 0; q < d; ++q) st += utt[q]; s_start = st; s_L = utt[d]; }
  __syncthreads();
  int start = s_start, L = s_L;
  for (int e = c; e < MAXL * MAXL; e += 256) {
    int p = e / MAXL, q = e % MAXL;
    if (p < L && q < L) {
      size_t idx = (size_t)b * (MAXL * MAXL) + e;
      Ablk[idx] = Ablk[idx] * dinv[m * NUTT + start + p] * dinv[m * NUTT + start + q];
    }
  }
}

// ---------------- 7) normalize cross links ----------------
__global__ __launch_bounds__(256) void k_normC(const float* __restrict__ crossv,
                                               const float* __restrict__ dinv,
                                               float* __restrict__ crossn) {
  int r = blockIdx.x * 256 + threadIdx.x;
  if (r >= 3 * NUTT) return;
  int p = r / NUTT, i = r % NUTT;
  int m0 = (p == 2) ? 1 : 0;
  int m1 = (p == 0) ? 1 : 2;
  crossn[r] = crossv[r] * dinv[m0 * NUTT + i] * dinv[m1 * NUTT + i];
}

// ---------------- 8) WMMA GEMM with fused epilogue ----------------
// C[M,256] = A[M,ktot] @ Wc (col-major f16). A split at k=256: A0 (k<256), A1 (k>=256).
// Each wave computes a 16x64 output tile (4 accumulators). K loop is software-
// pipelined x2: step s+1's 10 b128 loads are issued before step s's 4 WMMAs so
// loads overlap matrix ops and waits become partial.
// mode 0: out = relu(acc + bias[col])                          -> outF (f32) + outH (f16)
// mode 1: out = relu(theta*acc + (1-theta)*(0.9*hi + 0.1*h0))  -> outF (f32)
__global__ __launch_bounds__(256) void k_gemm(const _Float16* __restrict__ A0,
                                              const _Float16* __restrict__ A1,
                                              const _Float16* __restrict__ Wc,
                                              int ktot, int mode, float theta,
                                              const float* __restrict__ bias,
                                              const float* __restrict__ hi,
                                              const float* __restrict__ h0,
                                              float* __restrict__ outF,
                                              _Float16* __restrict__ outH) {
  const int lane = threadIdx.x & 31;
  const int wave = threadIdx.x >> 5;
  const int nG = CDIM / 64;                  // 4 column groups of 64
  const int ntiles = (MP / 16) * nG;         // 563 * 4 = 2252
  int tile = blockIdx.x * 8 + wave;          // wave-uniform predicate; EXEC stays full
  if (tile >= ntiles) return;
  const int tm  = (tile / nG) * 16;
  const int tng = (tile % nG) * 64;
  const int row  = tm + (lane & 15);
  const int koff = (lane >> 4) << 3;         // 0 or 8 (K sub-pattern per ISA A/B layout)

  const _Float16* rowA0 = A0 + (size_t)row * 256 + koff;
  const _Float16* rowA1 = A1 + (size_t)row * 256 + koff;
  // per-lane B column base: column tng + (lane&15); tiles j offset by 16 columns
  const _Float16* bbase = Wc + (size_t)(tng + (lane & 15)) * ktot + koff;

  v8f acc[4] = {};
  const int nsteps = ktot >> 5;              // 8 (fc) or 16 (layers); always even

#define LOAD_STEP(S, AL, AH, BL, BH)                                          \
  {                                                                           \
    int k_ = (S) << 5;                                                        \
    const _Float16* ab_ = (k_ < 256) ? (rowA0 + k_) : (rowA1 + (k_ - 256));   \
    AL = *(const half8*)ab_;                                                  \
    AH = *(const half8*)(ab_ + 16);                                           \
    const _Float16* bb_ = bbase + k_;                                         \
    _Pragma("unroll")                                                         \
    for (int j_ = 0; j_ < 4; ++j_) {                                          \
      BL[j_] = *(const half8*)(bb_ + (size_t)j_ * 16 * ktot);                 \
      BH[j_] = *(const half8*)(bb_ + (size_t)j_ * 16 * ktot + 16);            \
    }                                                                         \
  }

#define WMMA_STEP(AL, AH, BL, BH)                                             \
  {                                                                           \
    v16h af_ = __builtin_shufflevector(AL, AH,                                \
        0,1,2,3,4,5,6,7,8,9,10,11,12,13,14,15);                               \
    _Pragma("unroll")                                                         \
    for (int j_ = 0; j_ < 4; ++j_) {                                          \
      v16h bf_ = __builtin_shufflevector(BL[j_], BH[j_],                      \
          0,1,2,3,4,5,6,7,8,9,10,11,12,13,14,15);                             \
      acc[j_] = __builtin_amdgcn_wmma_f32_16x16x32_f16(false, af_, false,     \
          bf_, (short)0, acc[j_], false, false);                              \
    }                                                                         \
  }

  half8 alo0, ahi0, blo0[4], bhi0[4];
  half8 alo1, ahi1, blo1[4], bhi1[4];
  LOAD_STEP(0, alo0, ahi0, blo0, bhi0);
  for (int s = 0; s < nsteps; s += 2) {
    LOAD_STEP(s + 1, alo1, ahi1, blo1, bhi1);       // prefetch odd step
    WMMA_STEP(alo0, ahi0, blo0, bhi0);              // consume even step
    if (s + 2 < nsteps)
      LOAD_STEP(s + 2, alo0, ahi0, blo0, bhi0);     // prefetch next even step
    WMMA_STEP(alo1, ahi1, blo1, bhi1);              // consume odd step
  }
#undef LOAD_STEP
#undef WMMA_STEP

  const int rbase = tm + ((lane >> 4) << 3);
  #pragma unroll
  for (int j = 0; j < 4; ++j) {
    const int col = tng + j * 16 + (lane & 15);
    #pragma unroll
    for (int r = 0; r < 8; ++r) {
      int orow = rbase + r;
      if (orow < N3) {
        size_t idx = (size_t)orow * CDIM + col;
        float res;
        if (mode == 0) {
          res = acc[j][r] + bias[col];
        } else {
          res = theta * acc[j][r] + (1.0f - theta) * (0.9f * hi[idx] + 0.1f * h0[idx]);
        }
        res = fmaxf(res, 0.0f);
        outF[idx] = res;
        if (mode == 0) outH[idx] = (_Float16)res;
      }
    }
  }
}

// ---------------- 9) block SpMM: hi = A_norm @ h (f32 + f16 copy) ----------------
__global__ __launch_bounds__(256) void k_spmm(const float* __restrict__ h,
                                              const float* __restrict__ Ablk,
                                              const float* __restrict__ crossn,
                                              const int* __restrict__ utt,
                                              float* __restrict__ hi,
                                              _Float16* __restrict__ hih) {
  __shared__ float sh[MAXL * CDIM];     // 30KB
  __shared__ float sA[MAXL * MAXL];
  __shared__ int s_start, s_L;
  int b = blockIdx.x, m = b / NDIAL, d = b % NDIAL, c = threadIdx.x;
  if (c == 0) { int st = 0; for (int q = 0; q < d; ++q) st += utt[q]; s_start = st; s_L = utt[d]; }
  __syncthreads();
  int start = s_start, L = s_L;
  for (int q = 0; q < L; ++q)
    sh[q * CDIM + c] = h[((size_t)(m * NUTT + start + q)) * CDIM + c];
  for (int e = c; e < MAXL * MAXL; e += 256)
    sA[e] = Ablk[(size_t)b * (MAXL * MAXL) + e];
  __syncthreads();
  for (int p = 0; p < L; ++p) {
    int i = start + p;
    float s = 0.0f;
    for (int q = 0; q < L; ++q) s += sA[p * MAXL + q] * sh[q * CDIM + c];
    float c0, c1; int n0, n1;
    if (m == 0)      { c0 = crossn[0 * NUTT + i]; n0 = 1; c1 = crossn[1 * NUTT + i]; n1 = 2; }
    else if (m == 1) { c0 = crossn[0 * NUTT + i]; n0 = 0; c1 = crossn[2 * NUTT + i]; n1 = 2; }
    else             { c0 = crossn[1 * NUTT + i]; n0 = 0; c1 = crossn[2 * NUTT + i]; n1 = 1; }
    s += c0 * h[((size_t)(n0 * NUTT + i)) * CDIM + c]
       + c1 * h[((size_t)(n1 * NUTT + i)) * CDIM + c];
    size_t idx = ((size_t)(m * NUTT + i)) * CDIM + c;
    hi[idx] = s;
    hih[idx] = (_Float16)s;
  }
}

// ---------------- 10) output gather: [x_a|h_a|x_v|h_v|x_t|h_t] per utterance ----------------
__global__ __launch_bounds__(256) void k_out(const float* __restrict__ xf,
                                             const float* __restrict__ h,
                                             float* __restrict__ out) {
  int e = blockIdx.x * 256 + threadIdx.x;
  const int total = NUTT * 6 * CDIM;
  if (e >= total) return;
  int i = e / (6 * CDIM);
  int j = e % (6 * CDIM);
  int m = j / (2 * CDIM);
  int jj = j % (2 * CDIM);
  size_t src = ((size_t)(m * NUTT + i)) * CDIM + (jj % CDIM);
  out[e] = (jj < CDIM) ? xf[src] : h[src];
}

extern "C" void kernel_launch(void* const* d_in, const int* in_sizes, int n_in,
                              void* d_out, int out_size, void* d_ws, size_t ws_size,
                              hipStream_t stream) {
  (void)in_sizes; (void)n_in; (void)out_size; (void)ws_size;
  const float* a        = (const float*)d_in[0];
  const float* v        = (const float*)d_in[1];
  const float* t        = (const float*)d_in[2];
  const float* speaker  = (const float*)d_in[3];
  const int*   utt      = (const int*)  d_in[4];
  const float* spk_tab  = (const float*)d_in[5];
  const float* fc_w     = (const float*)d_in[6];
  const float* fc_b     = (const float*)d_in[7];
  const float* conv_w   = (const float*)d_in[8];
  float* out = (float*)d_out;

  char* ws = (char*)d_ws;
  size_t o = 0;
  auto take = [&](size_t bytes) -> char* {
    char* p = ws + o;
    o += (bytes + 255) & ~(size_t)255;
    return p;
  };
  float*    xf     = (float*)   take((size_t)N3 * CDIM * 4);
  float*    hats   = (float*)   take((size_t)N3 * CDIM * 4);   // reused as `hi` after A is built
  _Float16* Xh     = (_Float16*)take((size_t)MP * CDIM * 2);   // reused as `hih` after fc
  float*    h0f    = (float*)   take((size_t)N3 * CDIM * 4);
  _Float16* h0h    = (_Float16*)take((size_t)MP * CDIM * 2);
  float*    hcur   = (float*)   take((size_t)N3 * CDIM * 4);
  float*    Ablk   = (float*)   take((size_t)3 * NDIAL * MAXL * MAXL * 4);
  float*    crossv = (float*)   take((size_t)3 * NUTT * 4);
  float*    crossn = (float*)   take((size_t)3 * NUTT * 4);
  float*    dinv   = (float*)   take((size_t)N3 * 4);
  _Float16* fcWc   = (_Float16*)take((size_t)256 * 256 * 2);
  _Float16* convWc = (_Float16*)take((size_t)NLAYER * 512 * 256 * 2);
  float*    hi     = hats;   // alias (hats dead after graph construction)
  _Float16* hih    = Xh;     // alias (Xh dead after fc GEMM)

  const int gemm_blocks = ((MP / 16) * (CDIM / 64) + 7) / 8;   // ceil(2252/8) = 282

  k_wconv <<<4096, 256, 0, stream>>>(fc_w, conv_w, fcWc, convWc);
  k_pre   <<<NUTT, 256, 0, stream>>>(a, v, t, speaker, utt, spk_tab, xf, hats, Xh);
  k_cross <<<NUTT, 256, 0, stream>>>(hats, crossv);
  k_blocks<<<3 * NDIAL, 256, 0, stream>>>(hats, utt, Ablk);
  k_dinv  <<<(N3 + 255) / 256, 256, 0, stream>>>(Ablk, crossv, utt, dinv);
  k_normA <<<3 * NDIAL, 256, 0, stream>>>(Ablk, dinv, utt);
  k_normC <<<(3 * NUTT + 255) / 256, 256, 0, stream>>>(crossv, dinv, crossn);

  // fc: h0 = relu(x @ fc_w + fc_b)
  k_gemm<<<gemm_blocks, 256, 0, stream>>>(Xh, Xh, fcWc, 256, 0, 0.0f,
                                          fc_b, nullptr, nullptr, h0f, h0h);

  for (int l = 0; l < NLAYER; ++l) {
    float theta = logf(0.5f / (float)(l + 1) + 1.0f);
    const float* hin = (l == 0) ? h0f : hcur;
    k_spmm<<<3 * NDIAL, 256, 0, stream>>>(hin, Ablk, crossn, utt, hi, hih);
    k_gemm<<<gemm_blocks, 256, 0, stream>>>(hih, h0h, convWc + (size_t)l * 512 * 256,
                                            512, 1, theta, nullptr, hi, h0f, hcur, nullptr);
  }

  k_out<<<(NUTT * 6 * CDIM + 255) / 256, 256, 0, stream>>>(xf, hcur, out);
}
